// LORA_Attention_43800076485448
// MI455X (gfx1250) — compile-verified, matrix operand order checked by reading, not run on hardware
//
#include <hip/hip_runtime.h>
#include <hip/hip_bf16.h>

typedef __bf16 bf16_t;
typedef __attribute__((ext_vector_type(16))) __bf16 v16bf;
typedef __attribute__((ext_vector_type(8)))  float  v8f;
typedef __attribute__((ext_vector_type(4)))  unsigned int v4u;

#define BB 8
#define NN 1024
#define CC 768
#define HH 12
#define HD 64
#define D3 2304
#define M_TOT (BB * NN)           // 8192 tokens
static __device__ constexpr float SCALE = 0.125f;   // 64^-0.5

// ---------------------------------------------------------------------------
// Fragment helpers.
// 16x32 bf16 A/B fragment from a ROW-MAJOR tile (B is loaded as rows of B^T):
//   lane 0-15 : row = lane,    K = {0..7, 16..23}
//   lane 16-31: row = lane-16, K = {8..15, 24..31}
// Each lane pulls two contiguous 16-byte chunks.
// ---------------------------------------------------------------------------
union FragCvt { v16bf bf; v4u u[2]; };

__device__ __forceinline__ v16bf load_frag_rm(const bf16_t* src, int ld, int lane) {
    int row = lane & 15;
    int kb  = (lane >> 4) << 3;        // 0 or 8
    const bf16_t* p = src + (size_t)row * ld + kb;
    FragCvt f;
    f.u[0] = *(const v4u*)(p);
    f.u[1] = *(const v4u*)(p + 16);
    return f.bf;
}

__device__ __forceinline__ v8f wmma_bf16(v16bf a, v16bf b, v8f c) {
    // D = A(16x32) * B(32x16) + C, f32 accumulate
    return __builtin_amdgcn_wmma_f32_16x16x32_bf16(
        /*neg_a=*/false, a, /*neg_b=*/false, b,
        /*c_mod=*/(short)0, c, /*reuse_a=*/false, /*reuse_b=*/false);
}

// C/D 16x16 f32 layout: vgpr r, lanes 0-15 -> (M=r, N=lane);
// lanes 16-31 -> (M=r+8, N=lane-16).

// ---------------------------------------------------------------------------
// Prep kernels: fold rank-8 LoRA into dense weights, convert to bf16.
// ---------------------------------------------------------------------------
__global__ void fuse_qkv_w_kernel(const float* __restrict__ w,
                                  const float* __restrict__ qa, const float* __restrict__ qb,
                                  const float* __restrict__ ka, const float* __restrict__ kb,
                                  const float* __restrict__ va, const float* __restrict__ vb,
                                  bf16_t* __restrict__ out) {
    int idx = blockIdx.x * blockDim.x + threadIdx.x;
    if (idx >= D3 * CC) return;
    int d = idx / CC, c = idx % CC;
    const float* A; const float* Bm; int dl;
    if (d < CC)            { A = qa; Bm = qb; dl = d; }
    else if (d < 2 * CC)   { A = ka; Bm = kb; dl = d - CC; }
    else                   { A = va; Bm = vb; dl = d - 2 * CC; }
    float s = w[idx];
#pragma unroll
    for (int r = 0; r < 8; ++r) s += Bm[dl * 8 + r] * A[r * CC + c];
    out[idx] = (bf16_t)s;
}

__global__ void fuse_proj_w_kernel(const float* __restrict__ w,
                                   const float* __restrict__ oa, const float* __restrict__ ob,
                                   bf16_t* __restrict__ out) {
    int idx = blockIdx.x * blockDim.x + threadIdx.x;
    if (idx >= CC * CC) return;
    int d = idx / CC, c = idx % CC;
    float s = w[idx];
#pragma unroll
    for (int r = 0; r < 8; ++r) s += ob[d * 8 + r] * oa[r * CC + c];
    out[idx] = (bf16_t)s;
}

__global__ void to_bf16_kernel(const float* __restrict__ x, bf16_t* __restrict__ y, int n) {
    int i = blockIdx.x * blockDim.x + threadIdx.x;
    if (i < n) y[i] = (bf16_t)x[i];
}

// ---------------------------------------------------------------------------
// QKV GEMM: qkv[8192, 2304] = X[8192,768] * Weff^T + bias.
// One wave per 64x32 tile; 4x2 WMMA accumulators (64 VGPRs); K-step 32.
// 8 WMMAs per 6 fragment loads; ~130 VGPRs total -> no spills.
// Epilogue scatters q,k -> [B,H,N,HD] and v -> TRANSPOSED [B,H,HD,N] bf16.
// A 32-col tile never straddles q/k/v sections (768 % 32 == 0).
// ---------------------------------------------------------------------------
__global__ void gemm_qkv_kernel(const bf16_t* __restrict__ X, const bf16_t* __restrict__ W,
                                const float* __restrict__ bias,
                                bf16_t* __restrict__ q, bf16_t* __restrict__ k,
                                bf16_t* __restrict__ vt) {
    int lane = threadIdx.x;
    int m0 = blockIdx.x * 64;
    int n0 = blockIdx.y * 32;
    v8f acc[4][2] = {};
    for (int kk = 0; kk < CC; kk += 32) {
        v16bf b0 = load_frag_rm(W + (size_t)n0 * CC + kk, CC, lane);
        v16bf b1 = load_frag_rm(W + (size_t)(n0 + 16) * CC + kk, CC, lane);
#pragma unroll
        for (int i = 0; i < 4; ++i) {
            v16bf a = load_frag_rm(X + (size_t)(m0 + i * 16) * CC + kk, CC, lane);
            acc[i][0] = wmma_bf16(a, b0, acc[i][0]);
            acc[i][1] = wmma_bf16(a, b1, acc[i][1]);
        }
    }
    int n = lane & 15, mo = (lane >> 4) << 3;
#pragma unroll
    for (int i = 0; i < 4; ++i) {
#pragma unroll
        for (int j = 0; j < 2; ++j) {
            v8f a = acc[i][j];
            int col = n0 + j * 16 + n;
            float bv = bias[col];
            int sec = col / CC, f = col % CC;
            int h = f >> 6, d = f & 63;
#pragma unroll
            for (int r = 0; r < 8; ++r) {
                int tok = m0 + i * 16 + mo + r;
                int b = tok >> 10, t = tok & (NN - 1);
                bf16_t val = (bf16_t)(a[r] + bv);
                size_t bh = (size_t)(b * HH + h);
                if (sec == 0)       q[(bh * NN + t) * HD + d] = val;
                else if (sec == 1)  k[(bh * NN + t) * HD + d] = val;
                else                vt[(bh * HD + d) * NN + t] = val;   // transposed V
            }
        }
    }
}

// ---------------------------------------------------------------------------
// Attention: one wave per (b,h, 16-query tile). 3 passes:
//  1) S = Q K^T (WMMA) -> per-row max via tiny LDS reduction
//  2) recompute S, P = exp(S*scale - max) -> bf16 P tile in LDS, row sums
//  3) O = P V (A-frags from LDS, B-frags from V^T), normalize, store [B,N,C]
// ---------------------------------------------------------------------------
__global__ void attn_kernel(const bf16_t* __restrict__ Q, const bf16_t* __restrict__ K,
                            const bf16_t* __restrict__ Vt, bf16_t* __restrict__ O) {
    __shared__ __align__(16) bf16_t Pl[16 * NN];   // 32 KB bf16 P tile
    __shared__ float red[32 * 8];
    __shared__ float rmax[16];
    __shared__ float rsum[16];

    int lane = threadIdx.x;
    int qt = blockIdx.x;                  // 0..63 (query tile)
    int bh = blockIdx.y;                  // 0..95
    int b = bh / HH, h = bh % HH;

    const bf16_t* Qb = Q + ((size_t)bh * NN + qt * 16) * HD;
    const bf16_t* Kb = K + (size_t)bh * NN * HD;
    const bf16_t* Vb = Vt + (size_t)bh * HD * NN;

    v16bf aq0 = load_frag_rm(Qb + 0, HD, lane);   // K = 0..31
    v16bf aq1 = load_frag_rm(Qb + 32, HD, lane);  // K = 32..63

    int n = lane & 15;
    int mo = (lane >> 4) << 3;

    // ---- pass 1: row max ----
    float mx[8];
#pragma unroll
    for (int r = 0; r < 8; ++r) mx[r] = -1e30f;
    for (int j = 0; j < NN / 16; ++j) {
        v16bf bk0 = load_frag_rm(Kb + (size_t)(j * 16) * HD + 0, HD, lane);
        v16bf bk1 = load_frag_rm(Kb + (size_t)(j * 16) * HD + 32, HD, lane);
        v8f acc = {};
        acc = wmma_bf16(aq0, bk0, acc);
        acc = wmma_bf16(aq1, bk1, acc);
#pragma unroll
        for (int r = 0; r < 8; ++r) mx[r] = fmaxf(mx[r], acc[r] * SCALE);
    }
#pragma unroll
    for (int r = 0; r < 8; ++r) red[lane * 8 + r] = mx[r];
    __syncthreads();
    if (lane < 16) {
        int r = lane & 7;
        int l0 = (lane < 8) ? 0 : 16;     // rows 0-7 held by lanes 0-15; 8-15 by 16-31
        float m = -1e30f;
        for (int l = 0; l < 16; ++l) m = fmaxf(m, red[(l0 + l) * 8 + r]);
        rmax[lane] = m;
    }
    __syncthreads();

    // ---- pass 2: exp + P tile + row sums ----
    float rm[8], sm[8];
#pragma unroll
    for (int r = 0; r < 8; ++r) { rm[r] = rmax[mo + r]; sm[r] = 0.f; }
    for (int j = 0; j < NN / 16; ++j) {
        v16bf bk0 = load_frag_rm(Kb + (size_t)(j * 16) * HD + 0, HD, lane);
        v16bf bk1 = load_frag_rm(Kb + (size_t)(j * 16) * HD + 32, HD, lane);
        v8f acc = {};
        acc = wmma_bf16(aq0, bk0, acc);
        acc = wmma_bf16(aq1, bk1, acc);
#pragma unroll
        for (int r = 0; r < 8; ++r) {
            float p = __expf(acc[r] * SCALE - rm[r]);
            sm[r] += p;
            Pl[(mo + r) * NN + j * 16 + n] = (bf16_t)p;
        }
    }
    __syncthreads();
#pragma unroll
    for (int r = 0; r < 8; ++r) red[lane * 8 + r] = sm[r];
    __syncthreads();
    if (lane < 16) {
        int r = lane & 7;
        int l0 = (lane < 8) ? 0 : 16;
        float s = 0.f;
        for (int l = 0; l < 16; ++l) s += red[(l0 + l) * 8 + r];
        rsum[lane] = s;
    }
    __syncthreads();

    // ---- pass 3: O = P * V ----
    v8f accO[4] = {};
    for (int kk = 0; kk < NN; kk += 32) {
        v16bf ap = load_frag_rm((const bf16_t*)&Pl[kk], NN, lane);
#pragma unroll
        for (int t = 0; t < 4; ++t) {
            v16bf bv = load_frag_rm(Vb + (size_t)(t * 16) * NN + kk, NN, lane);
            accO[t] = wmma_bf16(ap, bv, accO[t]);
        }
    }
    float inv[8];
#pragma unroll
    for (int r = 0; r < 8; ++r) inv[r] = 1.0f / rsum[mo + r];
#pragma unroll
    for (int t = 0; t < 4; ++t) {
#pragma unroll
        for (int r = 0; r < 8; ++r) {
            size_t row = (size_t)b * NN + qt * 16 + mo + r;    // token index
            O[row * CC + h * HD + t * 16 + n] = (bf16_t)(accO[t][r] * inv[r]);
        }
    }
}

// ---------------------------------------------------------------------------
// Output projection: out[8192,768] = O * Wproj_eff^T + b_proj (f32 out).
// One wave per 64x32 tile, 4x2 accumulators.
// ---------------------------------------------------------------------------
__global__ void gemm_out_kernel(const bf16_t* __restrict__ X, const bf16_t* __restrict__ W,
                                const float* __restrict__ bias, float* __restrict__ out) {
    int lane = threadIdx.x;
    int m0 = blockIdx.x * 64;
    int n0 = blockIdx.y * 32;
    v8f acc[4][2] = {};
    for (int kk = 0; kk < CC; kk += 32) {
        v16bf b0 = load_frag_rm(W + (size_t)n0 * CC + kk, CC, lane);
        v16bf b1 = load_frag_rm(W + (size_t)(n0 + 16) * CC + kk, CC, lane);
#pragma unroll
        for (int i = 0; i < 4; ++i) {
            v16bf a = load_frag_rm(X + (size_t)(m0 + i * 16) * CC + kk, CC, lane);
            acc[i][0] = wmma_bf16(a, b0, acc[i][0]);
            acc[i][1] = wmma_bf16(a, b1, acc[i][1]);
        }
    }
    int n = lane & 15, mo = (lane >> 4) << 3;
#pragma unroll
    for (int i = 0; i < 4; ++i) {
#pragma unroll
        for (int j = 0; j < 2; ++j) {
            v8f a = acc[i][j];
            int col = n0 + j * 16 + n;
            float bv = bias[col];
#pragma unroll
            for (int r = 0; r < 8; ++r) {
                int tok = m0 + i * 16 + mo + r;
                out[(size_t)tok * CC + col] = a[r] + bv;
            }
        }
    }
}

// ---------------------------------------------------------------------------
extern "C" void kernel_launch(void* const* d_in, const int* in_sizes, int n_in,
                              void* d_out, int out_size, void* d_ws, size_t ws_size,
                              hipStream_t stream) {
    const float* x      = (const float*)d_in[0];
    const float* w_qkv  = (const float*)d_in[1];
    const float* b_qkv  = (const float*)d_in[2];
    const float* w_proj = (const float*)d_in[3];
    const float* b_proj = (const float*)d_in[4];
    const float* q_a = (const float*)d_in[5];
    const float* q_b = (const float*)d_in[6];
    const float* k_a = (const float*)d_in[7];
    const float* k_b = (const float*)d_in[8];
    const float* v_a = (const float*)d_in[9];
    const float* v_b = (const float*)d_in[10];
    const float* o_a = (const float*)d_in[11];
    const float* o_b = (const float*)d_in[12];
    float* out = (float*)d_out;

    // workspace carve-up (bf16 buffers), 256B aligned
    size_t off = 0;
    auto carve = [&](size_t bytes) {
        void* p = (char*)d_ws + off;
        off += (bytes + 255) & ~(size_t)255;
        return p;
    };
    bf16_t* xb   = (bf16_t*)carve((size_t)M_TOT * CC * 2);     // x bf16
    bf16_t* wq   = (bf16_t*)carve((size_t)D3 * CC * 2);        // fused QKV weight
    bf16_t* wp   = (bf16_t*)carve((size_t)CC * CC * 2);        // fused proj weight
    bf16_t* qbuf = (bf16_t*)carve((size_t)BB * HH * NN * HD * 2);
    bf16_t* kbuf = (bf16_t*)carve((size_t)BB * HH * NN * HD * 2);
    bf16_t* vtb  = (bf16_t*)carve((size_t)BB * HH * HD * NN * 2);
    bf16_t* obuf = (bf16_t*)carve((size_t)M_TOT * CC * 2);
    (void)ws_size; (void)n_in; (void)in_sizes; (void)out_size;

    // 1) convert x, fold LoRA into weights
    {
        int n = M_TOT * CC;
        to_bf16_kernel<<<(n + 255) / 256, 256, 0, stream>>>(x, xb, n);
        int nq = D3 * CC;
        fuse_qkv_w_kernel<<<(nq + 255) / 256, 256, 0, stream>>>(
            w_qkv, q_a, q_b, k_a, k_b, v_a, v_b, wq);
        int np = CC * CC;
        fuse_proj_w_kernel<<<(np + 255) / 256, 256, 0, stream>>>(w_proj, o_a, o_b, wp);
    }
    // 2) QKV projection (writes Q,K and transposed V)
    gemm_qkv_kernel<<<dim3(M_TOT / 64, D3 / 32), 32, 0, stream>>>(
        xb, wq, b_qkv, qbuf, kbuf, vtb);
    // 3) attention
    attn_kernel<<<dim3(NN / 16, BB * HH), 32, 0, stream>>>(qbuf, kbuf, vtb, obuf);
    // 4) output projection
    gemm_out_kernel<<<dim3(M_TOT / 64, CC / 32), 32, 0, stream>>>(obuf, wp, b_proj, out);
}
